// Attention_53927609368990
// MI455X (gfx1250) — matrix operand order repaired and checked by reading, not compiled, not run on hardware
//
#include <hip/hip_runtime.h>
#include <hip/hip_bf16.h>
#include <stdint.h>

// ---------------------------------------------------------------------------
// CDNA5 (gfx1250) shifted-window attention, bf16 WMMA everywhere.
// ---------------------------------------------------------------------------

typedef __attribute__((ext_vector_type(16))) __bf16 v16bf;
typedef __attribute__((ext_vector_type(8)))  float  v8f;

union FragB16 {
  uint4 q[2];   // 32 bytes = 16 bf16
  v16bf v;
};

#define WMMA_BF16(A, B, C) \
  __builtin_amdgcn_wmma_f32_16x16x32_bf16(false, (A), false, (B), (short)0, (C), false, false)

__device__ __forceinline__ unsigned short f2bf(float f) {
  unsigned int u = __float_as_uint(f);
  u += 0x7FFFu + ((u >> 16) & 1u);          // round-to-nearest-even
  return (unsigned short)(u >> 16);
}
__device__ __forceinline__ float bf2f(unsigned short h) {
  return __uint_as_float(((unsigned int)h) << 16);
}

#define NPIX   50176   // 16*56*56
#define CCH    384
#define NWH    12288   // 16*12*64 window-heads
#define DKS    0.17677669529663687f   // 32^-0.5

// ---------------------------------------------------------------------------
// K0: convert pointwise weights to bf16 ([O][I] row-major, K contiguous)
// ---------------------------------------------------------------------------
__global__ __launch_bounds__(256) void prep_weights(
    const float* __restrict__ wqkv, const float* __restrict__ wout,
    unsigned short* __restrict__ wqb, unsigned short* __restrict__ wob) {
  int i = blockIdx.x * 256 + threadIdx.x;
  if (i < 1152 * 384) wqb[i] = f2bf(wqkv[i]);
  if (i < 384 * 384)  wob[i] = f2bf(wout[i]);
}

// ---------------------------------------------------------------------------
// K1: roll(-3,-3) + depthwise 3x3 (zero pad) -> xb bf16 [pixel][channel]
// ---------------------------------------------------------------------------
__global__ __launch_bounds__(256) void dwconv_qkv(
    const float* __restrict__ x, const float* __restrict__ dw,
    unsigned short* __restrict__ xb) {
  int idx = blockIdx.x * 256 + threadIdx.x;
  if (idx >= NPIX * CCH) return;
  int c = idx % CCH;
  int pix = idx / CCH;
  int xx = pix % 56; int t = pix / 56;
  int y = t % 56; int b = t / 56;
  const float* wp = dw + c * 9;
  const float* xp = x + (size_t)(b * CCH + c) * (56 * 56);
  float acc = 0.f;
#pragma unroll
  for (int dy = -1; dy <= 1; ++dy) {
    int yy = y + dy;
    if (yy < 0 || yy >= 56) continue;
    int ys = yy + 3; if (ys >= 56) ys -= 56;          // roll shift = -3
#pragma unroll
    for (int dx = -1; dx <= 1; ++dx) {
      int x2 = xx + dx;
      if (x2 < 0 || x2 >= 56) continue;
      int xs = x2 + 3; if (xs >= 56) xs -= 56;
      acc += wp[(dy + 1) * 3 + (dx + 1)] * xp[ys * 56 + xs];
    }
  }
  xb[(size_t)pix * CCH + c] = f2bf(acc);
}

// ---------------------------------------------------------------------------
// K2: GEMM1  qkv[pix][1152] = xb[pix][384] @ W^T + bias, windowed epilogue.
// Block = 8 waves, 128(M) x 128(N) tile; wave = 32(M) x 64(N).
// ---------------------------------------------------------------------------
__global__ __launch_bounds__(256) void gemm_qkv(
    const unsigned short* __restrict__ xb, const unsigned short* __restrict__ wqb,
    const float* __restrict__ bias,
    unsigned short* __restrict__ qw, unsigned short* __restrict__ kw,
    unsigned short* __restrict__ vtw) {
  const int NB = 1152 / 128;                   // 9
  int bidN = blockIdx.x % NB;
  int bidM = blockIdx.x / NB;
  int tid = threadIdx.x;
  int wave = tid >> 5, lane = tid & 31;
  int l15 = lane & 15, hi = (lane >> 4) & 1;
  int wm = wave & 3, wn = wave >> 2;
  int M0 = bidM * 128 + wm * 32;
  int N0 = bidN * 128 + wn * 64;

  v8f zero8 = {0.f, 0.f, 0.f, 0.f, 0.f, 0.f, 0.f, 0.f};
  v8f acc[2][4];
#pragma unroll
  for (int i = 0; i < 2; ++i)
#pragma unroll
    for (int j = 0; j < 4; ++j) acc[i][j] = zero8;

  int kbA = hi ? 8 : 0;     // A lanes 16-31 hold K = 8..15 / 24..31
  int kbB = hi ? 8 : 0;     // B lanes 16-31 hold K = 8..15 / 24..31
  const unsigned short* arow0 = xb + (size_t)(M0 + l15) * 384;
  const unsigned short* arow1 = xb + (size_t)(M0 + 16 + l15) * 384;
  const unsigned short* brow[4];
#pragma unroll
  for (int ns = 0; ns < 4; ++ns)
    brow[ns] = wqb + (size_t)(N0 + ns * 16 + l15) * 384;

  for (int kk = 0; kk < 384; kk += 32) {
    FragB16 a[2], b[4];
    a[0].q[0] = *(const uint4*)(arow0 + kk + kbA);
    a[0].q[1] = *(const uint4*)(arow0 + kk + kbA + 16);
    a[1].q[0] = *(const uint4*)(arow1 + kk + kbA);
    a[1].q[1] = *(const uint4*)(arow1 + kk + kbA + 16);
#pragma unroll
    for (int ns = 0; ns < 4; ++ns) {
      const unsigned short* p = brow[ns] + kk + kbB;
      b[ns].q[0] = *(const uint4*)p;
      b[ns].q[1] = *(const uint4*)(p + 16);
    }
#pragma unroll
    for (int ms = 0; ms < 2; ++ms)
#pragma unroll
      for (int ns = 0; ns < 4; ++ns)
        acc[ms][ns] = WMMA_BF16(a[ms].v, b[ns].v, acc[ms][ns]);
  }

  // epilogue: bias, scale q, scatter into window layout
  int sec[4], hh[4], dd[4]; float bcol[4];
#pragma unroll
  for (int ns = 0; ns < 4; ++ns) {
    int c = N0 + ns * 16 + l15;
    bcol[ns] = bias[c];
    sec[ns] = c / 384; int ch = c % 384; hh[ns] = ch >> 5; dd[ns] = ch & 31;
  }
#pragma unroll
  for (int ms = 0; ms < 2; ++ms) {
#pragma unroll
    for (int v = 0; v < 8; ++v) {
      int row = M0 + ms * 16 + hi * 8 + v;
      int b = row / 3136; int rem = row - b * 3136;
      int y = rem / 56; int xx = rem - y * 56;
      int wy = y / 7, sy = y - wy * 7;
      int wx = xx / 7, sx = xx - wx * 7;
      int w = wy * 8 + wx;
      int t = sy * 7 + sx;
#pragma unroll
      for (int ns = 0; ns < 4; ++ns) {
        float val = acc[ms][ns][v] + bcol[ns];
        size_t wh = (size_t)((b * 12 + hh[ns]) * 64 + w);
        if (sec[ns] == 0)      qw[wh * 1568 + t * 32 + dd[ns]] = f2bf(val * DKS);
        else if (sec[ns] == 1) kw[wh * 1568 + t * 32 + dd[ns]] = f2bf(val);
        else                   vtw[wh * 2048 + dd[ns] * 64 + t] = f2bf(val);
      }
    }
  }
}

// ---------------------------------------------------------------------------
// K3: window attention. 1 block (4 waves) per window-head.
// dots(49x49 padded 64x64) = q @ k^T, softmax(+bias+mask), out = p @ v.
// ---------------------------------------------------------------------------
__global__ __launch_bounds__(128) void attn_win(
    const unsigned short* __restrict__ qw, const unsigned short* __restrict__ kw,
    const unsigned short* __restrict__ vtw, const float* __restrict__ pos,
    unsigned short* __restrict__ aw) {
  __shared__ float spos[169];
  __shared__ __align__(16) unsigned short sp[64 * 64];   // probs, bf16

  int wh = blockIdx.x;
  int tid = threadIdx.x;
  int wave = tid >> 5, lane = tid & 31;
  int l15 = lane & 15, hi = (lane >> 4) & 1;
  if (tid < 169) spos[tid] = pos[tid];
  __syncthreads();

  int w = wh & 63;
  int h = (wh >> 6) % 12;
  int b = wh / (64 * 12);
  int wy = w >> 3, wx = w & 7;
  int mwy = (wy == 7), mwx = (wx == 7);

  const unsigned short* qb = qw + (size_t)wh * 1568;
  const unsigned short* kb = kw + (size_t)wh * 1568;
  const unsigned short* vb = vtw + (size_t)wh * 2048;

  // A fragment: this wave's 16 q rows (tokens), K = 32 = head_dim
  FragB16 aq;
  {
    int t = wave * 16 + l15;
    if (t < 49) {
      const unsigned short* p = qb + t * 32 + (hi ? 8 : 0);
      aq.q[0] = *(const uint4*)p;
      aq.q[1] = *(const uint4*)(p + 16);
    } else {
      aq.q[0] = make_uint4(0, 0, 0, 0);
      aq.q[1] = make_uint4(0, 0, 0, 0);
    }
  }
  v8f zero8 = {0.f, 0.f, 0.f, 0.f, 0.f, 0.f, 0.f, 0.f};
  v8f dots[4];
#pragma unroll
  for (int cn = 0; cn < 4; ++cn) {
    FragB16 bk;
    int t = cn * 16 + l15;
    if (t < 49) {
      const unsigned short* p = kb + t * 32 + (hi ? 8 : 0);
      bk.q[0] = *(const uint4*)p;
      bk.q[1] = *(const uint4*)(p + 16);
    } else {
      bk.q[0] = make_uint4(0, 0, 0, 0);
      bk.q[1] = make_uint4(0, 0, 0, 0);
    }
    dots[cn] = WMMA_BF16(aq.v, bk.v, zero8);
  }

  // bias + shift mask
  float vals[4][8];
#pragma unroll
  for (int cn = 0; cn < 4; ++cn) {
    int j = cn * 16 + l15;
    int vj = (j < 49);
    int syj = vj ? (j / 7) : 0;
    int sxj = vj ? (j % 7) : 0;
#pragma unroll
    for (int v = 0; v < 8; ++v) {
      int i = wave * 16 + hi * 8 + v;
      float val;
      if (i < 49 && vj) {
        int syi = i / 7, sxi = i - syi * 7;
        val = dots[cn][v] + spos[(syj - syi + 6) * 13 + (sxj - sxi + 6)];
        int masked = (mwy && ((syi >= 4) != (syj >= 4))) ||
                     (mwx && ((sxi >= 4) != (sxj >= 4)));
        if (masked) val = -1e30f;
      } else {
        val = (i < 49) ? -1e30f : 0.0f;   // pad cols -> 0 prob; pad rows harmless
      }
      vals[cn][v] = val;
    }
  }

  // row softmax (row spread over 16 lanes x 4 column tiles)
#pragma unroll
  for (int v = 0; v < 8; ++v) {
    float m = fmaxf(fmaxf(vals[0][v], vals[1][v]), fmaxf(vals[2][v], vals[3][v]));
#pragma unroll
    for (int off = 1; off < 16; off <<= 1) m = fmaxf(m, __shfl_xor(m, off, 16));
    float s = 0.f;
#pragma unroll
    for (int cn = 0; cn < 4; ++cn) {
      vals[cn][v] = __expf(vals[cn][v] - m);
      s += vals[cn][v];
    }
#pragma unroll
    for (int off = 1; off < 16; off <<= 1) s += __shfl_xor(s, off, 16);
    float inv = 1.0f / s;
    int i = wave * 16 + hi * 8 + v;
#pragma unroll
    for (int cn = 0; cn < 4; ++cn)
      sp[i * 64 + cn * 16 + l15] = f2bf(vals[cn][v] * inv);
  }
  __syncthreads();

  // out = p(64x64) @ v(64x32): A from LDS (row-major p), B = vtw
  v8f o[2] = {zero8, zero8};
#pragma unroll
  for (int ks = 0; ks < 2; ++ks) {
    FragB16 pa;
    const unsigned short* pr = sp + (wave * 16 + l15) * 64 + ks * 32 + (hi ? 8 : 0);
    pa.q[0] = *(const uint4*)pr;
    pa.q[1] = *(const uint4*)(pr + 16);
#pragma unroll
    for (int nd = 0; nd < 2; ++nd) {
      FragB16 bv;
      int d = nd * 16 + l15;
      const unsigned short* p = vb + d * 64 + ks * 32 + (hi ? 8 : 0);
      bv.q[0] = *(const uint4*)p;
      bv.q[1] = *(const uint4*)(p + 16);
      o[nd] = WMMA_BF16(pa.v, bv.v, o[nd]);
    }
  }

  // scatter back to channel-last image layout (window merge)
#pragma unroll
  for (int nd = 0; nd < 2; ++nd) {
    int d = nd * 16 + l15;
    int c = h * 32 + d;
#pragma unroll
    for (int v = 0; v < 8; ++v) {
      int t = wave * 16 + hi * 8 + v;
      if (t < 49) {
        int sy = t / 7, sx = t - sy * 7;
        int y = wy * 7 + sy, x = wx * 7 + sx;
        aw[(((size_t)b * 56 + y) * 56 + x) * 384 + c] = f2bf(o[nd][v]);
      }
    }
  }
}

// ---------------------------------------------------------------------------
// K4: roll(+3,+3) + depthwise 3x3 -> yb bf16 [pixel][channel]
// ---------------------------------------------------------------------------
__global__ __launch_bounds__(256) void dwconv_out(
    const unsigned short* __restrict__ aw, const float* __restrict__ dw,
    unsigned short* __restrict__ yb) {
  int idx = blockIdx.x * 256 + threadIdx.x;
  if (idx >= NPIX * CCH) return;
  int c = idx % CCH;
  int pix = idx / CCH;
  int xx = pix % 56; int t = pix / 56;
  int y = t % 56; int b = t / 56;
  const float* wp = dw + c * 9;
  float acc = 0.f;
#pragma unroll
  for (int dy = -1; dy <= 1; ++dy) {
    int yy = y + dy;
    if (yy < 0 || yy >= 56) continue;
    int ys = yy - 3; if (ys < 0) ys += 56;            // roll shift = +3
#pragma unroll
    for (int dx = -1; dx <= 1; ++dx) {
      int x2 = xx + dx;
      if (x2 < 0 || x2 >= 56) continue;
      int xs = x2 - 3; if (xs < 0) xs += 56;
      acc += wp[(dy + 1) * 3 + (dx + 1)] *
             bf2f(aw[(((size_t)b * 56 + ys) * 56 + xs) * 384 + c]);
    }
  }
  yb[(size_t)pix * CCH + c] = f2bf(acc);
}

// ---------------------------------------------------------------------------
// K5: GEMM2  out[pix][384] = yb[pix][384] @ W^T + bias -> fp32 NCHW
// ---------------------------------------------------------------------------
__global__ __launch_bounds__(256) void gemm_out(
    const unsigned short* __restrict__ yb, const unsigned short* __restrict__ wob,
    const float* __restrict__ bias, float* __restrict__ out) {
  const int NB = 384 / 128;                   // 3
  int bidN = blockIdx.x % NB;
  int bidM = blockIdx.x / NB;
  int tid = threadIdx.x;
  int wave = tid >> 5, lane = tid & 31;
  int l15 = lane & 15, hi = (lane >> 4) & 1;
  int wm = wave & 3, wn = wave >> 2;
  int M0 = bidM * 128 + wm * 32;
  int N0 = bidN * 128 + wn * 64;

  v8f zero8 = {0.f, 0.f, 0.f, 0.f, 0.f, 0.f, 0.f, 0.f};
  v8f acc[2][4];
#pragma unroll
  for (int i = 0; i < 2; ++i)
#pragma unroll
    for (int j = 0; j < 4; ++j) acc[i][j] = zero8;

  int kbA = hi ? 8 : 0;
  int kbB = hi ? 8 : 0;
  const unsigned short* arow0 = yb + (size_t)(M0 + l15) * 384;
  const unsigned short* arow1 = yb + (size_t)(M0 + 16 + l15) * 384;
  const unsigned short* brow[4];
#pragma unroll
  for (int ns = 0; ns < 4; ++ns)
    brow[ns] = wob + (size_t)(N0 + ns * 16 + l15) * 384;

  for (int kk = 0; kk < 384; kk += 32) {
    FragB16 a[2], b[4];
    a[0].q[0] = *(const uint4*)(arow0 + kk + kbA);
    a[0].q[1] = *(const uint4*)(arow0 + kk + kbA + 16);
    a[1].q[0] = *(const uint4*)(arow1 + kk + kbA);
    a[1].q[1] = *(const uint4*)(arow1 + kk + kbA + 16);
#pragma unroll
    for (int ns = 0; ns < 4; ++ns) {
      const unsigned short* p = brow[ns] + kk + kbB;
      b[ns].q[0] = *(const uint4*)p;
      b[ns].q[1] = *(const uint4*)(p + 16);
    }
#pragma unroll
    for (int ms = 0; ms < 2; ++ms)
#pragma unroll
      for (int ns = 0; ns < 4; ++ns)
        acc[ms][ns] = WMMA_BF16(a[ms].v, b[ns].v, acc[ms][ns]);
  }

  float bcol[4]; int cc[4];
#pragma unroll
  for (int ns = 0; ns < 4; ++ns) {
    cc[ns] = N0 + ns * 16 + l15;
    bcol[ns] = bias[cc[ns]];
  }
#pragma unroll
  for (int ms = 0; ms < 2; ++ms) {
#pragma unroll
    for (int v = 0; v < 8; ++v) {
      int row = M0 + ms * 16 + hi * 8 + v;
      int b = row / 3136; int rem = row - b * 3136;
      int y = rem / 56; int xx = rem - y * 56;
#pragma unroll
      for (int ns = 0; ns < 4; ++ns)
        out[(((size_t)b * 384 + cc[ns]) * 56 + y) * 56 + xx] = acc[ms][ns][v] + bcol[ns];
    }
  }
}

// ---------------------------------------------------------------------------
extern "C" void kernel_launch(void* const* d_in, const int* in_sizes, int n_in,
                              void* d_out, int out_size, void* d_ws, size_t ws_size,
                              hipStream_t stream) {
  (void)in_sizes; (void)n_in; (void)out_size; (void)ws_size;
  const float* x       = (const float*)d_in[0];
  const float* qkv_dw  = (const float*)d_in[1];
  const float* qkv_pw  = (const float*)d_in[2];
  const float* qkv_pb  = (const float*)d_in[3];
  const float* out_dw  = (const float*)d_in[4];
  const float* out_pw  = (const float*)d_in[5];
  const float* out_pb  = (const float*)d_in[6];
  const float* pos_emb = (const float*)d_in[7];
  float* out = (float*)d_out;

  char* ws = (char*)d_ws;
  const size_t XB = (size_t)NPIX * CCH * 2;        // 38,535,168  (xb, reused as aw)
  const size_t QW = (size_t)NWH * 1568 * 2;        // 38,535,168  (qw, reused as yb)
  const size_t KW = QW;                            // kw
  const size_t VT = (size_t)NWH * 2048 * 2;        // 50,331,648  (v transposed, padded)
  unsigned short* xb  = (unsigned short*)ws;
  unsigned short* qw  = (unsigned short*)(ws + XB);
  unsigned short* kw  = (unsigned short*)(ws + XB + QW);
  unsigned short* vtw = (unsigned short*)(ws + XB + QW + KW);
  unsigned short* wqb = (unsigned short*)(ws + XB + QW + KW + VT);
  unsigned short* wob = wqb + 1152 * 384;
  unsigned short* aw  = xb;   // xb dead after gemm_qkv
  unsigned short* yb  = qw;   // qw dead after attn_win

  hipMemsetAsync(vtw, 0, VT, stream);   // zero token padding for attn@v B-matrix
  prep_weights<<<(1152 * 384 + 255) / 256, 256, 0, stream>>>(qkv_pw, out_pw, wqb, wob);
  dwconv_qkv<<<(NPIX * CCH) / 256, 256, 0, stream>>>(x, qkv_dw, xb);
  gemm_qkv<<<392 * 9, 256, 0, stream>>>(xb, wqb, qkv_pb, qw, kw, vtw);
  attn_win<<<NWH, 128, 0, stream>>>(qw, kw, vtw, pos_emb, aw);
  dwconv_out<<<(NPIX * CCH) / 256, 256, 0, stream>>>(aw, out_dw, yb);
  gemm_out<<<392 * 3, 256, 0, stream>>>(yb, wob, out_pb, out);
}
